// CNN_88098369175803
// MI455X (gfx1250) — matrix operand order repaired and checked by reading, not compiled, not run on hardware
//
#include <hip/hip_runtime.h>
#include <hip/hip_bf16.h>

// ---------------------------------------------------------------------------
// Problem constants (from reference)
// ---------------------------------------------------------------------------
#define BB   64
#define CC   64
#define TT   8192
#define HH   10
#define OO   2
#define HP   16          // H padded to WMMA tile
#define VTH  0.5f
#define TAUc 0.25f
#define A_MEM 0.1f       // DT * TAU_MEM_INV
#define C_SYN 0.8f       // 1 - DT * TAU_SYN_INV

typedef __attribute__((ext_vector_type(16))) _Float16 v16h;
typedef __attribute__((ext_vector_type(8)))  float    v8f;

// ---------------------------------------------------------------------------
// Kernel 1: front GEMM  y[b, m(0..15), t] = sum_c w[m,c] * eeg[b,c,t]
// NOTE: b_front is intentionally omitted — it is a per-channel constant over
// (B,T) and the immediately following td_bn subtracts the per-channel mean
// over (B,T), so the bias cancels exactly (variance unchanged). Dropping it
// removes a predicated-load epilogue with zero numerical effect.
// 256-thread block = 8 waves; each wave owns one 16(M) x 16(T) tile.
// K = 64 -> two v_wmma_f32_16x16x32_f16, fully unrolled.
// Fragment layouts per CDNA5 ISA 7.12.2; all loads branch-free
// (clamped address + value select).
// ---------------------------------------------------------------------------
__global__ __launch_bounds__(256) void gemm_front_wmma(
    const float* __restrict__ eeg,   // (B, C, T) f32
    const float* __restrict__ w,     // (H, C)    f32
    float* __restrict__ out)         // (B, 16, T) f32
{
  const int wave = threadIdx.x >> 5;                 // 0..7
  const int t0   = (blockIdx.x * 8 + wave) * 16;
  const int b    = blockIdx.y;
  const int lane = threadIdx.x & 31;
  const int half = lane >> 4;      // lane group 0..1
  const int lm   = lane & 15;      // M for A-frag, N for B/C-frag
  const int lmc  = (lm < HH) ? lm : (HH - 1);        // clamped A row

  v8f c = {};
#pragma unroll
  for (int ks = 0; ks < CC; ks += 32) {
    v16h a, bm;
#pragma unroll
    for (int j = 0; j < 16; ++j) {
      // A (16x32, 16-bit): VGPR v=j/2; K = (v>=4?16:0) + half*8 + (v%4)*2 + (j&1)
      const int v  = j >> 1;
      const int ka = ((v >> 2) << 4) + half * 8 + ((v & 3) << 1) + (j & 1);
      const float wv = w[lmc * CC + ks + ka];        // always in-bounds
      a[j] = (_Float16)((lm < HH) ? wv : 0.0f);      // v_cndmask, no branch
      // B (32x16, 16-bit): same K order as A: K = (j>=8?16:0) + half*8 + (j&7)
      const int kb = ((j >> 3) << 4) + half * 8 + (j & 7);
      bm[j] = (_Float16)eeg[((size_t)b * CC + (ks + kb)) * TT + t0 + lm];
    }
    c = __builtin_amdgcn_wmma_f32_16x16x32_f16(false, a, false, bm,
                                               (short)0, c, false, false);
  }
#pragma unroll
  for (int r = 0; r < 8; ++r) {
    const int m = r + 8 * half;                 // C/D layout: VGPR r -> M=r+8*half
    out[((size_t)b * HP + m) * TT + t0 + lm] = c[r];
  }
}

// ---------------------------------------------------------------------------
// Kernel 2: generic padded HxH GEMM  y[b,m,t] = sum_h w[m,h] * x[b,h,t]
// K = 16 zero-padded to 32 -> one WMMA per tile. B-fragment rows >=16 are
// clamped to row 15, which is guaranteed zero (spike rows 10..15 are zero),
// so no select is needed on the B side at all.
// ---------------------------------------------------------------------------
__global__ __launch_bounds__(256) void gemm_h_wmma(
    const _Float16* __restrict__ x,  // (B, 16, T) f16 spikes, rows 10..15 zero
    const float* __restrict__ w,     // (H, H) f32 row-major
    float* __restrict__ out)         // (B, 16, T) f32
{
  const int wave = threadIdx.x >> 5;
  const int t0   = (blockIdx.x * 8 + wave) * 16;
  const int b    = blockIdx.y;
  const int lane = threadIdx.x & 31;
  const int half = lane >> 4;
  const int lm   = lane & 15;
  const int lmc  = (lm < HH) ? lm : (HH - 1);

  v16h a, bm;
#pragma unroll
  for (int j = 0; j < 16; ++j) {
    const int v  = j >> 1;
    const int ka = ((v >> 2) << 4) + half * 8 + ((v & 3) << 1) + (j & 1);
    const int kac = (ka < HH) ? ka : (HH - 1);
    const float wv = w[lmc * HH + kac];              // always in-bounds
    a[j] = (_Float16)((lm < HH && ka < HH) ? wv : 0.0f);
    const int kb  = ((j >> 3) << 4) + half * 8 + (j & 7);
    const int kbc = (kb < HP) ? kb : (HP - 1);       // row 15 is zero
    bm[j] = x[((size_t)b * HP + kbc) * TT + t0 + lm];
  }
  v8f c = {};
  c = __builtin_amdgcn_wmma_f32_16x16x32_f16(false, a, false, bm,
                                             (short)0, c, false, false);
#pragma unroll
  for (int r = 0; r < 8; ++r) {
    const int m = r + 8 * half;
    out[((size_t)b * HP + m) * TT + t0 + lm] = c[r];
  }
}

// ---------------------------------------------------------------------------
// Kernel 3: td_bn statistics. One block per channel m, reduce over B*T.
// stats[m] = mean, stats[16+m] = rsqrt(var + eps)
// ---------------------------------------------------------------------------
__global__ __launch_bounds__(256) void bn_stats_kernel(
    const float* __restrict__ pre,   // (B, 16, T)
    float* __restrict__ stats)       // (32)
{
  const int m   = blockIdx.x;
  const int tid = threadIdx.x;
  const size_t N = (size_t)BB * TT;
  float s = 0.0f, s2 = 0.0f;
  for (size_t i = tid; i < N; i += 256) {
    const size_t b = i >> 13;          // /T
    const size_t t = i & (TT - 1);
    const float v = pre[(b * HP + m) * TT + t];
    s += v; s2 += v * v;
  }
  __shared__ float sh0[256], sh1[256];
  sh0[tid] = s; sh1[tid] = s2;
  __syncthreads();
  for (int off = 128; off > 0; off >>= 1) {
    if (tid < off) { sh0[tid] += sh0[tid + off]; sh1[tid] += sh1[tid + off]; }
    __syncthreads();
  }
  if (tid == 0) {
    const float mean = sh0[0] / (float)N;
    const float var  = sh1[0] / (float)N - mean * mean;
    stats[m]      = mean;
    stats[16 + m] = rsqrtf(var + 1e-5f);
  }
}

// ---------------------------------------------------------------------------
// Kernel 4: fused BN-affine + LIF time scan (hard spike forward).
// Thread per (b, m). stats==nullptr -> plain lif. layout_tbh selects output
// layout: 0 -> (B,16,T), 1 -> (T, B*16) (coalesced feed for LSNN).
// Chunked with a software prefetch (global_prefetch_b8) since the scan is
// latency-bound, not bandwidth-bound.
// ---------------------------------------------------------------------------
__global__ __launch_bounds__(256) void bn_lif_kernel(
    const float* __restrict__ pre,     // (B, 16, T)
    const float* __restrict__ stats,   // (32) or nullptr
    const float* __restrict__ gamma,   // (H) or nullptr
    const float* __restrict__ beta,    // (H) or nullptr
    _Float16* __restrict__ out,
    int layout_tbh)
{
  const int tid = blockIdx.x * blockDim.x + threadIdx.x;  // 0..1023
  const int b = tid >> 4, m = tid & 15;
  float scale, shift;
  if (stats != nullptr) {
    if (m < HH) {
      scale = VTH * gamma[m] * stats[16 + m];
      shift = beta[m] - scale * stats[m];
    } else { scale = 0.0f; shift = 0.0f; }
  } else {
    scale = (m < HH) ? 1.0f : 0.0f; shift = 0.0f;
  }
  const float* row = pre + ((size_t)b * HP + m) * TT;
  float u = 0.0f;
  for (int t0 = 0; t0 < TT; t0 += 64) {
    __builtin_prefetch(row + t0 + 128, 0, 0);   // global_prefetch_b8
#pragma unroll 8
    for (int t = t0; t < t0 + 64; ++t) {
      const float xi = scale * row[t] + shift;
      u = TAUc * u + xi;
      const float s = (u >= VTH) ? 1.0f : 0.0f;
      u *= (1.0f - s);
      if (layout_tbh) out[(size_t)t * (BB * HP) + b * HP + m] = (_Float16)s;
      else            out[((size_t)b * HP + m) * TT + t]      = (_Float16)s;
    }
  }
}

// ---------------------------------------------------------------------------
// Kernel 5: windowed attention. Thread per (b, t):
// aw[w] = sum_h q[b,h,t]*k[b,h,t+w-4] / H ;  av[b,h,t] = sum_w aw[w]*v[..]
// ---------------------------------------------------------------------------
__global__ __launch_bounds__(256) void attn_kernel(
    const _Float16* __restrict__ q,
    const _Float16* __restrict__ k,
    const _Float16* __restrict__ v,
    float* __restrict__ out)           // (B, 16, T)
{
  const int idx = blockIdx.x * 256 + threadIdx.x;  // b*T + t
  const int b = idx >> 13;
  const int t = idx & (TT - 1);
  float aw[9];
#pragma unroll
  for (int w = 0; w < 9; ++w) aw[w] = 0.0f;

  for (int h = 0; h < HH; ++h) {
    const size_t base = ((size_t)b * HP + h) * TT;
    const float qv = (float)q[base + t];
#pragma unroll
    for (int w = 0; w < 9; ++w) {
      const int tt = t + w - 4;
      if (tt >= 0 && tt < TT) aw[w] += qv * (float)k[base + tt];
    }
  }
#pragma unroll
  for (int w = 0; w < 9; ++w) aw[w] *= (1.0f / (float)HH);

  for (int h = 0; h < HH; ++h) {
    const size_t base = ((size_t)b * HP + h) * TT;
    float acc = 0.0f;
#pragma unroll
    for (int w = 0; w < 9; ++w) {
      const int tt = t + w - 4;
      if (tt >= 0 && tt < TT) acc += aw[w] * (float)v[base + tt];
    }
    out[base + t] = acc;
  }
  for (int h = HH; h < HP; ++h)
    out[((size_t)b * HP + h) * TT + t] = 0.0f;
}

// ---------------------------------------------------------------------------
// Kernel 6: LSNN recurrence + LI readout. Single block, thread = (b, h).
// State shared through LDS; exact carry ordering of the reference scan.
// ---------------------------------------------------------------------------
__global__ __launch_bounds__(1024) void lsnn_li_kernel(
    const _Float16* __restrict__ seq,  // (T, B*16) f16 spikes
    const float* __restrict__ w_in,    // (10,10)
    const float* __restrict__ w_rec,   // (10,10)
    const float* __restrict__ w_cls,   // (2,10)
    float* __restrict__ out)           // (B, 2)
{
  const int tid = threadIdx.x;         // 0..1023
  const int b = tid >> 4, h = tid & 15;

  __shared__ float xi_sh[BB * HP];
  __shared__ float z_sh[BB * HP];
  __shared__ float win_sh[HH * HH], wrec_sh[HH * HH], wcls_sh[OO * HH];
  if (tid < HH * HH) { win_sh[tid] = w_in[tid]; wrec_sh[tid] = w_rec[tid]; }
  if (tid < OO * HH) wcls_sh[tid] = w_cls[tid];
  z_sh[tid] = 0.0f;

  float vm = 0.0f, ii = 0.0f;          // LSNN membrane / synaptic current
  float vcls = 0.0f, icls = 0.0f, acc = 0.0f;
  __syncthreads();

  for (int t = 0; t < TT; ++t) {
    // decays from OLD state
    const float v_dec = vm + A_MEM * (ii - vm);
    const float z_new = (h < HH && v_dec >= VTH) ? 1.0f : 0.0f;
    vm = v_dec * (1.0f - z_new);

    xi_sh[tid] = (float)seq[(size_t)t * (BB * HP) + tid];
    __syncthreads();                   // xi ready; z_sh still holds z_{t-1}

    float sum = 0.0f;
    if (h < HH) {
      const int rb = b << 4;
#pragma unroll
      for (int hp = 0; hp < HH; ++hp)
        sum += xi_sh[rb + hp] * win_sh[h * HH + hp]
             + z_sh[rb + hp] * wrec_sh[h * HH + hp];
    }
    ii = ii * C_SYN + sum;
    __syncthreads();                   // all reads of z_{t-1} done
    z_sh[tid] = z_new;
    __syncthreads();                   // z_t published

    if (h < OO) {                      // LI readout consumes z_t
      const float vnew = vcls + A_MEM * (icls - vcls);
      acc += vnew;
      float dot = 0.0f;
      const int rb = b << 4;
#pragma unroll
      for (int hp = 0; hp < HH; ++hp) dot += z_sh[rb + hp] * wcls_sh[h * HH + hp];
      icls = icls * C_SYN + dot;
      vcls = vnew;
    }
  }
  if (h < OO) out[b * OO + h] = acc / (float)TT;
}

// ---------------------------------------------------------------------------
// Host-side pipeline
// ---------------------------------------------------------------------------
extern "C" void kernel_launch(void* const* d_in, const int* in_sizes, int n_in,
                              void* d_out, int out_size, void* d_ws, size_t ws_size,
                              hipStream_t stream) {
  (void)in_sizes; (void)n_in; (void)out_size; (void)ws_size;

  const float* beeg    = (const float*)d_in[2];   // (B,1,C,T) -> (B,C,T)
  const float* w_front = (const float*)d_in[4];
  // d_in[5] (b_front) intentionally unused: cancels exactly through td_bn.
  const float* gi      = (const float*)d_in[6];
  const float* bi      = (const float*)d_in[7];
  const float* wq      = (const float*)d_in[8];
  const float* gq      = (const float*)d_in[9];
  const float* bq      = (const float*)d_in[10];
  const float* wk      = (const float*)d_in[11];
  const float* gk      = (const float*)d_in[12];
  const float* bk      = (const float*)d_in[13];
  const float* wv      = (const float*)d_in[14];
  const float* gv      = (const float*)d_in[15];
  const float* bv      = (const float*)d_in[16];
  const float* w_cma   = (const float*)d_in[17];
  const float* gc      = (const float*)d_in[18];
  const float* bc      = (const float*)d_in[19];
  const float* w_in    = (const float*)d_in[20];
  const float* w_rec   = (const float*)d_in[21];
  const float* w_cls   = (const float*)d_in[22];

  // workspace layout
  const size_t F32_BYTES = (size_t)BB * HP * TT * sizeof(float);     // 32 MiB
  const size_t F16_BYTES = (size_t)BB * HP * TT * sizeof(_Float16);  // 16 MiB
  char* w8 = (char*)d_ws;
  float*     pre  = (float*)w8;
  _Float16*  xb   = (_Float16*)(w8 + F32_BYTES);
  _Float16*  qb   = (_Float16*)(w8 + F32_BYTES + 1 * F16_BYTES);
  _Float16*  kb   = (_Float16*)(w8 + F32_BYTES + 2 * F16_BYTES);
  _Float16*  vb   = (_Float16*)(w8 + F32_BYTES + 3 * F16_BYTES);
  _Float16*  ab   = (_Float16*)(w8 + F32_BYTES + 4 * F16_BYTES);
  _Float16*  seq  = (_Float16*)(w8 + F32_BYTES + 5 * F16_BYTES);     // (T,B*16)
  float*     stats= (float*)(w8 + F32_BYTES + 6 * F16_BYTES);

  const dim3 gemm_grid(TT / (16 * 8), BB);   // 8 waves (tiles) per block
  const dim3 gemm_blk(256);
  const dim3 lif_grid((BB * HP) / 256);
  const dim3 stat_grid(HP);

  // front: linear + bn + lif -> x
  gemm_front_wmma<<<gemm_grid, gemm_blk, 0, stream>>>(beeg, w_front, pre);
  bn_stats_kernel<<<stat_grid, 256, 0, stream>>>(pre, stats);
  bn_lif_kernel<<<lif_grid, 256, 0, stream>>>(pre, stats, gi, bi, xb, 0);

  // q / k / v branches
  gemm_h_wmma<<<gemm_grid, gemm_blk, 0, stream>>>(xb, wq, pre);
  bn_stats_kernel<<<stat_grid, 256, 0, stream>>>(pre, stats);
  bn_lif_kernel<<<lif_grid, 256, 0, stream>>>(pre, stats, gq, bq, qb, 0);

  gemm_h_wmma<<<gemm_grid, gemm_blk, 0, stream>>>(xb, wk, pre);
  bn_stats_kernel<<<stat_grid, 256, 0, stream>>>(pre, stats);
  bn_lif_kernel<<<lif_grid, 256, 0, stream>>>(pre, stats, gk, bk, kb, 0);

  gemm_h_wmma<<<gemm_grid, gemm_blk, 0, stream>>>(xb, wv, pre);
  bn_stats_kernel<<<stat_grid, 256, 0, stream>>>(pre, stats);
  bn_lif_kernel<<<lif_grid, 256, 0, stream>>>(pre, stats, gv, bv, vb, 0);

  // windowed attention -> plain lif -> cma linear -> bn -> lif (T-major out)
  attn_kernel<<<(BB * TT) / 256, 256, 0, stream>>>(qb, kb, vb, pre);
  bn_lif_kernel<<<lif_grid, 256, 0, stream>>>(pre, nullptr, nullptr, nullptr, ab, 0);
  gemm_h_wmma<<<gemm_grid, gemm_blk, 0, stream>>>(ab, w_cma, pre);
  bn_stats_kernel<<<stat_grid, 256, 0, stream>>>(pre, stats);
  bn_lif_kernel<<<lif_grid, 256, 0, stream>>>(pre, stats, gc, bc, seq, 1);

  // recurrent LSNN + LI readout
  lsnn_li_kernel<<<1, 1024, 0, stream>>>(seq, w_in, w_rec, w_cls, (float*)d_out);
}
